// Model_25915832664269
// MI455X (gfx1250) — matrix operand order repaired and checked by reading, not compiled, not run on hardware
//
#include <hip/hip_runtime.h>
#include <hip/hip_bf16.h>
#include <hip/hip_fp16.h>

// ---------------------------------------------------------------------------
// CDNA5 (gfx1250) implementation.
// Pipeline: edge-MLP (3->64->128->512, lrelu) -> fused segment-max
//           -> encoder head (515->512 lrelu, 512->1024)
//           -> optimal latent -> FoldingNet decoder (2x 3-layer MLP).
// Wide GEMMs: v_wmma_f32_16x16x32_f16, double-buffered LDS staged with
// global_load_async_to_lds_b128 (ASYNCcnt pipeline).
// ---------------------------------------------------------------------------

typedef __attribute__((ext_vector_type(16))) _Float16 v16h;
typedef __attribute__((ext_vector_type(8)))  float    v8f;

#define BM 128
#define BN 128
#define BK 32
#define LDA 40   // LDS stride (halves) for A tile rows  (80B = 5x16B aligned)
#define LDB 40   // LDS stride (halves) for B^T tile rows

// reference constants
#define BATCH  32
#define NCENT  64
#define GRIDN  45
#define GG     (GRIDN * GRIDN)   // 2025
#define RROWS  (BATCH * GG)      // 64800

// order-preserving float<->int map (involution) for atomicMax-based seg-max
__device__ __forceinline__ int enc_f32(float v) {
  int b = __float_as_int(v);
  return b >= 0 ? b : (b ^ 0x7fffffff);
}
__device__ __forceinline__ float dec_f32(int k) {
  int b = k >= 0 ? k : (k ^ 0x7fffffff);
  return __int_as_float(b);
}

// ---- gfx1250 async global->LDS copy (16B per lane), ASYNCcnt tracked ------
__device__ __forceinline__ void async_ld16(unsigned lds_addr, unsigned long long gaddr) {
  asm volatile("global_load_async_to_lds_b128 %0, %1, off"
               :: "v"(lds_addr), "v"(gaddr) : "memory");
}
__device__ __forceinline__ void wait_async_4() {
#if __has_builtin(__builtin_amdgcn_s_wait_asynccnt)
  __builtin_amdgcn_s_wait_asynccnt(4);
#else
  asm volatile("s_wait_asynccnt 0x4" ::: "memory");
#endif
}
__device__ __forceinline__ void wait_async_0() {
#if __has_builtin(__builtin_amdgcn_s_wait_asynccnt)
  __builtin_amdgcn_s_wait_asynccnt(0);
#else
  asm volatile("s_wait_asynccnt 0x0" ::: "memory");
#endif
}
__device__ __forceinline__ unsigned lds_off(const void* p) {
  return (unsigned)(size_t)p;  // flat LDS addr low 32 bits == LDS offset
}

// ---- WMMA fragment loaders (CDNA5 16-bit layouts, CDNA5 ISA) ----
// A (16x32, MxK): lane L<16 holds row M=L, halves 0..7 = K 0..7, 8..15 = K 16..23
//                 lane L+16 holds row M=L, halves 0..7 = K 8..15, 8..15 = K 24..31
__device__ __forceinline__ v16h frag_a(const _Float16* As /*16-row chunk base*/) {
  const int lane = threadIdx.x & 31;
  const int m    = lane & 15;
  const int kh   = (lane & 16) ? 8 : 0;
  const _Float16* p = As + m * LDA + kh;
  v16h a;
#pragma unroll
  for (int i = 0; i < 8; ++i) a[i] = p[i];
#pragma unroll
  for (int i = 0; i < 8; ++i) a[8 + i] = p[16 + i];
  return a;
}
// B (32x16, KxN) staged as B^T rows: BsT[n*LDB + k]
// lane L<16 holds col N=L, K=0..7,16..23 ; lane L+16 holds col N=L, K=8..15,24..31
__device__ __forceinline__ v16h frag_b(const _Float16* BsT /*16-col chunk base*/) {
  const int lane = threadIdx.x & 31;
  const int n    = lane & 15;
  const int k0   = (lane & 16) ? 8 : 0;
  const _Float16* p = BsT + n * LDB + k0;
  v16h b;
#pragma unroll
  for (int i = 0; i < 8; ++i) b[i] = p[i];
#pragma unroll
  for (int i = 0; i < 8; ++i) b[8 + i] = p[16 + i];
  return b;
}

// ---------------------------------------------------------------------------
// Templated WMMA GEMM:  C[M,N] = epi(A[M,K] @ B + bias), B passed TRANSPOSED
// as Bt[N][K] (row-major, ld = K).  Requires N % 128 == 0, K % 32 == 0.
// EPI: 0 none, 1 leaky-relu(0.2), 2 relu
// OUT: 0 f16 store, 1 f32 store, 2 segment-max scatter via atomicMax(enc)
// Double-buffered: async tile t+1 issued before computing tile t.
// ---------------------------------------------------------------------------
template <int EPI, int OUT>
__global__ __launch_bounds__(256) void wmma_gemm(
    const _Float16* __restrict__ Ain, const _Float16* __restrict__ Bt,
    const float* __restrict__ bias, void* __restrict__ Cout,
    const int* __restrict__ yidx, int M, int N, int K, int ldc) {
  __shared__ __align__(16) _Float16 As[2][BM * LDA];
  __shared__ __align__(16) _Float16 Bs[2][BN * LDB];

  const int tid = threadIdx.x;
  const int wid = tid >> 5;
  const int wm  = wid & 3;   // 4 wave-rows  -> 32 rows each
  const int wn  = wid >> 2;  // 2 wave-cols  -> 64 cols each
  const int m0  = blockIdx.x * BM;
  const int n0  = blockIdx.y * BN;

  // per-thread staging assignment: 32B of A + 32B of B per tile
  const int sr = tid >> 1;             // 0..127 : A row / Bt row in tile
  const int sc = (tid & 1) * 16;       // 0 or 16 halves within the 32-half line
  int gra = m0 + sr; if (gra >= M) gra = M - 1;  // clamp: rows >= M discarded later
  const _Float16* gA = Ain + (size_t)gra * K + sc;
  const _Float16* gB = Bt + (size_t)(n0 + sr) * K + sc;

  v8f acc[2][4];
#pragma unroll
  for (int i = 0; i < 2; ++i)
#pragma unroll
    for (int j = 0; j < 4; ++j)
#pragma unroll
      for (int r = 0; r < 8; ++r) acc[i][j][r] = 0.0f;

  // issue: 4 async b128 per thread (= 4 ASYNCcnt per wave) for one k-tile
  auto stage = [&](int k0, int buf) {
    const unsigned la = lds_off(&As[buf][sr * LDA + sc]);
    const unsigned lb = lds_off(&Bs[buf][sr * LDB + sc]);
    const unsigned long long pa = (unsigned long long)(gA + k0);
    const unsigned long long pb = (unsigned long long)(gB + k0);
    __builtin_prefetch(gA + k0 + 2 * BK, 0, 0);   // global_prefetch_b8 (speculative)
    async_ld16(la,      pa);
    async_ld16(la + 16, pa + 8);
    async_ld16(lb,      pb);
    async_ld16(lb + 16, pb + 8);
  };

  stage(0, 0);
  const int T = K / BK;
  for (int t = 0; t < T; ++t) {
    const int buf = t & 1;
    if (t + 1 < T) {
      stage((t + 1) * BK, buf ^ 1);
      wait_async_4();        // in-order: tile t complete, tile t+1 in flight
    } else {
      wait_async_0();
    }
    __syncthreads();         // tile-t LDS visible to all waves

    v16h a[2], b[4];
#pragma unroll
    for (int i = 0; i < 2; ++i) a[i] = frag_a(&As[buf][(wm * 32 + i * 16) * LDA]);
#pragma unroll
    for (int j = 0; j < 4; ++j) b[j] = frag_b(&Bs[buf][(wn * 64 + j * 16) * LDB]);
#pragma unroll
    for (int i = 0; i < 2; ++i)
#pragma unroll
      for (int j = 0; j < 4; ++j)
        acc[i][j] = __builtin_amdgcn_wmma_f32_16x16x32_f16(
            false, a[i], false, b[j], (short)0, acc[i][j], false, false);
    __syncthreads();         // retire buf before it is overwritten at t+2
  }

  // --- epilogue: C VGPR r holds row r (lanes 0-15) / row r+8 (lanes 16-31)
  const int lane   = tid & 31;
  const int colb   = lane & 15;
  const int rowoff = (lane & 16) ? 8 : 0;
#pragma unroll
  for (int i = 0; i < 2; ++i) {
#pragma unroll
    for (int j = 0; j < 4; ++j) {
      const int col = n0 + wn * 64 + j * 16 + colb;
      const float bv = bias ? bias[col] : 0.0f;
#pragma unroll
      for (int r = 0; r < 8; ++r) {
        const int row = m0 + wm * 32 + i * 16 + rowoff + r;
        if (row < M) {
          float v = acc[i][j][r] + bv;
          if (EPI == 1) v = v > 0.f ? v : 0.2f * v;
          if (EPI == 2) v = v > 0.f ? v : 0.f;
          if (OUT == 0) {
            ((_Float16*)Cout)[(size_t)row * ldc + col] = (_Float16)v;
          } else if (OUT == 1) {
            ((float*)Cout)[(size_t)row * ldc + col] = v;
          } else {
            const int seg = yidx[row];
            atomicMax(((int*)Cout) + (size_t)seg * ldc + col, enc_f32(v));
          }
        }
      }
    }
  }
}

// ---------------------------------------------------------------------------
// Small helper kernels
// ---------------------------------------------------------------------------
// fp32 W[Ks][N] -> f16 transposed + K-padded  Wt[N][Kp]  (pad rows = 0)
__global__ void convert_pad_tr_f16(const float* __restrict__ src,
                                   _Float16* __restrict__ dst, int Ks, int Kp, int N) {
  const int idx = blockIdx.x * blockDim.x + threadIdx.x;
  if (idx >= Kp * N) return;
  const int n = idx / Kp, k = idx - n * Kp;
  dst[idx] = (k < Ks) ? (_Float16)src[(size_t)k * N + n] : (_Float16)0.f;
}

__global__ void fill_i32(int* __restrict__ p, int val, int n) {
  const int idx = blockIdx.x * blockDim.x + threadIdx.x;
  if (idx < n) p[idx] = val;
}

// layer1: rel = pos[x] - pos[fps[y]];  h1 = lrelu(rel @ W1 + b1)  (K=3 -> VALU)
__global__ void edge_layer1(const float* __restrict__ pos, const int* __restrict__ fps_idx,
                            const int* __restrict__ x_idx, const int* __restrict__ y_idx,
                            const float* __restrict__ W1, const float* __restrict__ b1,
                            _Float16* __restrict__ h1, int E) {
  const int e = blockIdx.x * blockDim.x + threadIdx.x;
  if (e >= E) return;
  const int xi = x_idx[e];
  const int ci = fps_idx[y_idx[e]];
  const float rx = pos[3 * xi + 0] - pos[3 * ci + 0];
  const float ry = pos[3 * xi + 1] - pos[3 * ci + 1];
  const float rz = pos[3 * xi + 2] - pos[3 * ci + 2];
#pragma unroll
  for (int j = 0; j < 64; ++j) {
    float v = rx * W1[j] + ry * W1[64 + j] + rz * W1[128 + j] + b1[j];
    v = v > 0.f ? v : 0.2f * v;
    h1[(size_t)e * 64 + j] = (_Float16)v;
  }
}

// encIn[m, 0:512]=decode(featEnc), [512:515]=center xyz, [515:544]=0   (f16)
__global__ void build_encin(const int* __restrict__ featEnc, const float* __restrict__ pos,
                            const int* __restrict__ fps_idx, _Float16* __restrict__ encIn,
                            int M) {
  const int idx = blockIdx.x * blockDim.x + threadIdx.x;
  if (idx >= M * 544) return;
  const int m = idx / 544, c = idx - m * 544;
  float v;
  if (c < 512)       v = dec_f32(featEnc[(size_t)m * 512 + c]);
  else if (c < 515)  v = pos[3 * fps_idx[m] + (c - 512)];
  else               v = 0.f;
  encIn[idx] = (_Float16)v;
}

// opt[b,j] = sum_s mean/std / sum_s 1/std   over selected centers
__global__ void compute_opt(const float* __restrict__ z, const int* __restrict__ sel,
                            float* __restrict__ opt, int nsub) {
  const int j = threadIdx.x;   // 512
  const int b = blockIdx.x;    // BATCH
  float num = 0.f, den = 0.f;
  for (int s = 0; s < nsub; ++s) {
    const size_t row = (size_t)b * NCENT + sel[s];
    const float mean = z[row * 1024 + j];
    const float lv   = z[row * 1024 + 512 + j];
    const float istd = __expf(-0.5f * lv);  // 1/exp(0.5*logvar)
    num += mean * istd;
    den += istd;
  }
  opt[b * 512 + j] = num / den;
}

// decIn[row, 0:512]=code, [512]=lin[g%45], [513]=lin[g/45], rest 0   (f16)
__global__ void build_decin(const float* __restrict__ opt, _Float16* __restrict__ decIn) {
  const size_t idx = (size_t)blockIdx.x * blockDim.x + threadIdx.x;
  if (idx >= (size_t)RROWS * 544) return;
  const int row = (int)(idx / 544), c = (int)(idx - (size_t)row * 544);
  const int b = row / GG, g = row - b * GG;
  float v;
  if (c < 512)       v = opt[b * 512 + c];
  else if (c == 512) v = -0.3f + 0.6f * (float)(g % GRIDN) / 44.f;
  else if (c == 513) v = -0.3f + 0.6f * (float)(g / GRIDN) / 44.f;
  else               v = 0.f;
  decIn[idx] = (_Float16)v;
}

// out3[row,0:3] = X[row,0:512] @ W[512,3] + b ; tail: out3[R*3] = 0
__global__ void proj3(const _Float16* __restrict__ X, const float* __restrict__ W,
                      const float* __restrict__ b, float* __restrict__ out3,
                      int R, int write_tail) {
  const int row = blockIdx.x * blockDim.x + threadIdx.x;
  if (row == 0 && write_tail) out3[(size_t)R * 3] = 0.f;
  if (row >= R) return;
  const _Float16* x = X + (size_t)row * 512;
  float s0 = b[0], s1 = b[1], s2 = b[2];
  for (int k = 0; k < 512; ++k) {
    const float xv = (float)x[k];
    s0 += xv * W[k * 3 + 0];
    s1 += xv * W[k * 3 + 1];
    s2 += xv * W[k * 3 + 2];
  }
  out3[(size_t)row * 3 + 0] = s0;
  out3[(size_t)row * 3 + 1] = s1;
  out3[(size_t)row * 3 + 2] = s2;
}

// overwrite decIn cols 512..514 with f3 (code part stays) -> fold-2 input
__global__ void inject_f3(const float* __restrict__ f3, _Float16* __restrict__ decIn) {
  const int idx = blockIdx.x * blockDim.x + threadIdx.x;
  if (idx >= RROWS * 3) return;
  const int row = idx / 3, c = idx - row * 3;
  decIn[(size_t)row * 544 + 512 + c] = (_Float16)f3[idx];
}

// ---------------------------------------------------------------------------
extern "C" void kernel_launch(void* const* d_in, const int* in_sizes, int n_in,
                              void* d_out, int out_size, void* d_ws, size_t ws_size,
                              hipStream_t stream) {
  (void)n_in; (void)out_size; (void)ws_size;
  const float* pos     = (const float*)d_in[0];
  const int*   fps_idx = (const int*)d_in[2];
  const int*   x_idx   = (const int*)d_in[3];
  const int*   y_idx   = (const int*)d_in[4];
  const int*   sel_idx = (const int*)d_in[5];
  const float* W1 = (const float*)d_in[6];   const float* b1  = (const float*)d_in[7];
  const float* b2  = (const float*)d_in[9];  const float* b3  = (const float*)d_in[11];
  const float* be1 = (const float*)d_in[13]; const float* be2 = (const float*)d_in[15];
  const float* F1b1 = (const float*)d_in[17]; const float* F1b2 = (const float*)d_in[19];
  const float* F1W3 = (const float*)d_in[20]; const float* F1b3 = (const float*)d_in[21];
  const float* F2b1 = (const float*)d_in[23]; const float* F2b2 = (const float*)d_in[25];
  const float* F2W3 = (const float*)d_in[26]; const float* F2b3 = (const float*)d_in[27];

  const int E = in_sizes[3];   // number of edges
  const int M = in_sizes[2];   // 2048 centers
  const int NSUBc = in_sizes[5];

  // ---- workspace carving (256B aligned, edge buffers union'd with decIn) ----
  char* ws = (char*)d_ws;
  size_t off = 0;
  auto alloc = [&](size_t bytes) {
    size_t o = off; off = (off + bytes + 255) & ~(size_t)255; return o;
  };
  _Float16* wW2   = (_Float16*)(ws + alloc((size_t)64 * 128 * 2));    // [128][64]
  _Float16* wW3   = (_Float16*)(ws + alloc((size_t)128 * 512 * 2));   // [512][128]
  _Float16* wWe1  = (_Float16*)(ws + alloc((size_t)544 * 512 * 2));   // [512][544]
  _Float16* wWe2  = (_Float16*)(ws + alloc((size_t)512 * 1024 * 2));  // [1024][512]
  _Float16* wF1W1 = (_Float16*)(ws + alloc((size_t)544 * 512 * 2));
  _Float16* wF1W2 = (_Float16*)(ws + alloc((size_t)512 * 512 * 2));
  _Float16* wF2W1 = (_Float16*)(ws + alloc((size_t)544 * 512 * 2));
  _Float16* wF2W2 = (_Float16*)(ws + alloc((size_t)512 * 512 * 2));

  const size_t h1B = ((size_t)E * 64 * 2 + 255) & ~(size_t)255;
  const size_t edgeBytes = h1B + (size_t)E * 128 * 2;
  const size_t decBytes  = (size_t)RROWS * 544 * 2;
  const size_t uOff = alloc(edgeBytes > decBytes ? edgeBytes : decBytes);
  _Float16* h1    = (_Float16*)(ws + uOff);
  _Float16* h2    = (_Float16*)(ws + uOff + h1B);
  _Float16* decIn = (_Float16*)(ws + uOff);   // reused after edge stage retires

  int*      featEnc = (int*)     (ws + alloc((size_t)M * 512 * 4));
  _Float16* encIn   = (_Float16*)(ws + alloc((size_t)M * 544 * 2));
  _Float16* z1      = (_Float16*)(ws + alloc((size_t)M * 512 * 2));
  float*    zf      = (float*)   (ws + alloc((size_t)M * 1024 * 4));
  float*    opt     = (float*)   (ws + alloc((size_t)BATCH * 512 * 4));
  _Float16* f1      = (_Float16*)(ws + alloc((size_t)RROWS * 512 * 2));
  _Float16* f2      = (_Float16*)(ws + alloc((size_t)RROWS * 512 * 2));
  float*    f3      = (float*)   (ws + alloc((size_t)RROWS * 3 * 4));
  float*    outp    = (float*)d_out;

  auto cdiv = [](long long a, long long b) -> unsigned { return (unsigned)((a + b - 1) / b); };
  dim3 blk(256);

  // ---- weight conversion (fp32 [K][N] -> f16 transposed [N][Kp], K padded) ----
  convert_pad_tr_f16<<<cdiv(64 * 128, 256), blk, 0, stream>>>((const float*)d_in[8],  wW2,   64,  64,  128);
  convert_pad_tr_f16<<<cdiv(128 * 512, 256), blk, 0, stream>>>((const float*)d_in[10], wW3,  128, 128,  512);
  convert_pad_tr_f16<<<cdiv(544 * 512, 256), blk, 0, stream>>>((const float*)d_in[12], wWe1, 515, 544,  512);
  convert_pad_tr_f16<<<cdiv(512 * 1024, 256), blk, 0, stream>>>((const float*)d_in[14], wWe2, 512, 512, 1024);
  convert_pad_tr_f16<<<cdiv(544 * 512, 256), blk, 0, stream>>>((const float*)d_in[16], wF1W1, 514, 544, 512);
  convert_pad_tr_f16<<<cdiv(512 * 512, 256), blk, 0, stream>>>((const float*)d_in[18], wF1W2, 512, 512, 512);
  convert_pad_tr_f16<<<cdiv(544 * 512, 256), blk, 0, stream>>>((const float*)d_in[22], wF2W1, 515, 544, 512);
  convert_pad_tr_f16<<<cdiv(512 * 512, 256), blk, 0, stream>>>((const float*)d_in[24], wF2W2, 512, 512, 512);

  // ---- edge MLP + fused segment-max ----
  edge_layer1<<<cdiv(E, 256), blk, 0, stream>>>(pos, fps_idx, x_idx, y_idx, W1, b1, h1, E);
  fill_i32<<<cdiv((long long)M * 512, 256), blk, 0, stream>>>(featEnc, 0x807FFFFF /*enc(-inf)*/, M * 512);
  wmma_gemm<1, 0><<<dim3(cdiv(E, BM), 128 / BN), blk, 0, stream>>>(
      h1, wW2, b2, h2, nullptr, E, 128, 64, 128);
  wmma_gemm<1, 2><<<dim3(cdiv(E, BM), 512 / BN), blk, 0, stream>>>(
      h2, wW3, b3, featEnc, y_idx, E, 512, 128, 512);

  // ---- encoder head ----
  build_encin<<<cdiv((long long)M * 544, 256), blk, 0, stream>>>(featEnc, pos, fps_idx, encIn, M);
  wmma_gemm<1, 0><<<dim3(cdiv(M, BM), 512 / BN), blk, 0, stream>>>(
      encIn, wWe1, be1, z1, nullptr, M, 512, 544, 512);
  wmma_gemm<0, 1><<<dim3(cdiv(M, BM), 1024 / BN), blk, 0, stream>>>(
      z1, wWe2, be2, zf, nullptr, M, 1024, 512, 1024);
  compute_opt<<<dim3(BATCH), dim3(512), 0, stream>>>(zf, sel_idx, opt, NSUBc);

  // ---- FoldingNet decoder, fold 1 ----
  build_decin<<<cdiv((long long)RROWS * 544, 256), blk, 0, stream>>>(opt, decIn);
  wmma_gemm<2, 0><<<dim3(cdiv(RROWS, BM), 512 / BN), blk, 0, stream>>>(
      decIn, wF1W1, F1b1, f1, nullptr, RROWS, 512, 544, 512);
  wmma_gemm<2, 0><<<dim3(cdiv(RROWS, BM), 512 / BN), blk, 0, stream>>>(
      f1, wF1W2, F1b2, f2, nullptr, RROWS, 512, 512, 512);
  proj3<<<cdiv(RROWS, 256), blk, 0, stream>>>(f2, F1W3, F1b3, f3, RROWS, 0);

  // ---- fold 2 (code part of decIn is reused; only 3 coord cols replaced) ----
  inject_f3<<<cdiv((long long)RROWS * 3, 256), blk, 0, stream>>>(f3, decIn);
  wmma_gemm<2, 0><<<dim3(cdiv(RROWS, BM), 512 / BN), blk, 0, stream>>>(
      decIn, wF2W1, F2b1, f1, nullptr, RROWS, 512, 544, 512);
  wmma_gemm<2, 0><<<dim3(cdiv(RROWS, BM), 512 / BN), blk, 0, stream>>>(
      f1, wF2W2, F2b2, f2, nullptr, RROWS, 512, 512, 512);
  proj3<<<cdiv(RROWS, 256), blk, 0, stream>>>(f2, F2W3, F2b3, outp, RROWS, 1 /*tail zero*/);
}
